// Mamba1D_5866925326508
// MI455X (gfx1250) — matrix operand order repaired and checked by reading, not compiled, not run on hardware
//
#include <hip/hip_runtime.h>
#include <hip/hip_bf16.h>
#include <stdint.h>

typedef __attribute__((ext_vector_type(16))) _Float16 v16h;
typedef __attribute__((ext_vector_type(8)))  float    v8f;
typedef int v4ig __attribute__((vector_size(16)));   // GCC-style int4 (async-LDS builtin param type)

static constexpr int Bsz = 16, Lsz = 1024, Dsz = 256;
static constexpr int NST = 16, Kc = 4, NL = 4, Hh = 64;
static constexpr int DI  = 512, Rr = 16;
static constexpr int Mrows = Bsz * Lsz;          // 16384
static constexpr int NPROJ = Rr + 2 * NST;       // 48

struct alignas(16) U4 { uint32_t v[4]; };

// ---------------------------------------------------------------------------
// Elementwise input projection: h[b,l,d] = x[b,l]*w[d] + bias[d]
// ---------------------------------------------------------------------------
__global__ void k_input_proj(const float* __restrict__ x,
                             const float* __restrict__ w,
                             const float* __restrict__ bias,
                             float* __restrict__ h)
{
    int idx = blockIdx.x * blockDim.x + threadIdx.x;
    if (idx >= Mrows * Dsz) return;
    int d  = idx % Dsz;
    int ml = idx / Dsz;
    h[idx] = x[ml] * w[d] + bias[d];
}

// ---------------------------------------------------------------------------
// f32 -> f16 conversion (weights)
// ---------------------------------------------------------------------------
__global__ void k_f32_to_f16(const float* __restrict__ src,
                             _Float16* __restrict__ dst, int n)
{
    int i = blockIdx.x * blockDim.x + threadIdx.x;
    if (i < n) dst[i] = (_Float16)src[i];
}

// ---------------------------------------------------------------------------
// LayerNorm over D=256; one block per row; emits f16 for the WMMA GEMM.
// ---------------------------------------------------------------------------
__global__ __launch_bounds__(256)
void k_layernorm(const float* __restrict__ h,
                 const float* __restrict__ g,
                 const float* __restrict__ b,
                 _Float16* __restrict__ xn)
{
    __shared__ float red[256];
    const int row = blockIdx.x;
    const int t   = threadIdx.x;
    float v = h[(size_t)row * Dsz + t];
    red[t] = v; __syncthreads();
    #pragma unroll
    for (int s = 128; s > 0; s >>= 1) { if (t < s) red[t] += red[t + s]; __syncthreads(); }
    float mean = red[0] * (1.0f / Dsz);
    __syncthreads();
    float dv = v - mean;
    red[t] = dv * dv; __syncthreads();
    #pragma unroll
    for (int s = 128; s > 0; s >>= 1) { if (t < s) red[t] += red[t + s]; __syncthreads(); }
    float rstd = rsqrtf(red[0] * (1.0f / Dsz) + 1e-5f);
    xn[(size_t)row * Dsz + t] = (_Float16)(dv * rstd * g[t] + b[t]);
}

// ---------------------------------------------------------------------------
// LDS-tiled WMMA GEMM: C[M,N] = resid + A[M,K] * W[N,K]^T
// Block = 256 threads = 8 waves (2 M x 4 N), macro-tile 64(M) x 128(N),
// K stepped by 32, DOUBLE-BUFFERED LDS staged with async global->LDS copies
// (ASYNCcnt) when the toolchain exposes the gfx1250 builtin; synchronous
// b128 copies otherwise. Each wave computes 32x32 via 4 accumulators.
// Fragment layouts per CDNA5 ISA 7.12.2:
//   A 16x32 f16 : low lanes u0..3 <- K 0..7,  u4..7 <- K 16..23
//                 high lanes u0..3 <- K 8..15, u4..7 <- K 24..31
//   B 32x16 f16 : col = lane%16; same K split across lanes/u0..7 as A
//   C/D 16x16 f32: vgpr i -> row (half*8 + i), col = lane%16
// ---------------------------------------------------------------------------
__global__ __launch_bounds__(256)
void k_gemm_wmma_tiled(const _Float16* __restrict__ A, int lda,
                       const _Float16* __restrict__ W, int ldb,
                       float* __restrict__ C, int ldc,
                       const float* __restrict__ resid,
                       int Ktot)
{
    constexpr int MT = 64, NT = 128, KT = 32, LDSS = KT + 8;   // 40-half row stride
    __shared__ __align__(16) _Float16 As[2][MT * LDSS];
    __shared__ __align__(16) _Float16 Bs[2][NT * LDSS];

    const int t    = threadIdx.x;
    const int lane = t & 31;
    const int half = lane >> 4;
    const int lrow = lane & 15;
    const int w    = t >> 5;        // wave 0..7
    const int wm   = w >> 2;        // 0..1  (M direction)
    const int wn   = w & 3;         // 0..3  (N direction)
    const int mBase = blockIdx.y * MT;
    const int nBase = blockIdx.x * NT;

    v8f acc[2][2] = {};

    const int ar = t >> 2;          // 0..63   A tile row
    const int ac = (t & 3) * 8;     // half-col 0,8,16,24
    const int br = t >> 1;          // 0..127  B tile row
    const int bc = (t & 1) * 16;    // half-col 0,16

    // Stage one 32-deep k-chunk of A (64x32) and B (128x32) into buffer p.
    auto stage = [&](int kb, int p) {
        const _Float16* ga  = A + (size_t)(mBase + ar) * lda + kb + ac;
        const _Float16* gb0 = W + (size_t)(nBase + br) * ldb + kb + bc;
#if defined(__gfx1250__) && __has_builtin(__builtin_amdgcn_global_load_async_to_lds_b128)
        __builtin_amdgcn_global_load_async_to_lds_b128(
            (__attribute__((address_space(1))) v4ig*)ga,
            (__attribute__((address_space(3))) v4ig*)&As[p][ar * LDSS + ac], 0, 0);
        __builtin_amdgcn_global_load_async_to_lds_b128(
            (__attribute__((address_space(1))) v4ig*)gb0,
            (__attribute__((address_space(3))) v4ig*)&Bs[p][br * LDSS + bc], 0, 0);
        __builtin_amdgcn_global_load_async_to_lds_b128(
            (__attribute__((address_space(1))) v4ig*)(gb0 + 8),
            (__attribute__((address_space(3))) v4ig*)&Bs[p][br * LDSS + bc + 8], 0, 0);
#else
        *reinterpret_cast<U4*>(&As[p][ar * LDSS + ac])     = *reinterpret_cast<const U4*>(ga);
        *reinterpret_cast<U4*>(&Bs[p][br * LDSS + bc])     = *reinterpret_cast<const U4*>(gb0);
        *reinterpret_cast<U4*>(&Bs[p][br * LDSS + bc + 8]) = *reinterpret_cast<const U4*>(gb0 + 8);
#endif
    };
    auto wait_stage = [&]() {
#if defined(__gfx1250__) && __has_builtin(__builtin_amdgcn_global_load_async_to_lds_b128)
#if __has_builtin(__builtin_amdgcn_s_wait_asynccnt)
        __builtin_amdgcn_s_wait_asynccnt(0);
#else
        asm volatile("s_wait_asynccnt 0" ::: "memory");
#endif
#endif
    };

    stage(0, 0);
    int p = 0;
    for (int kb = 0; kb < Ktot; kb += KT, p ^= 1) {
        wait_stage();            // this wave's async writes for chunk kb landed
        __syncthreads();         // all waves' writes visible
        if (kb + KT < Ktot) stage(kb + KT, p ^ 1);   // prefetch next chunk

        union Frag { U4 q[2]; v16h v; };
        Frag a[2], b[2];
        #pragma unroll
        for (int mi = 0; mi < 2; ++mi) {
            int row = (wm * 32 + mi * 16 + lrow) * LDSS;
            a[mi].q[0] = *reinterpret_cast<const U4*>(&As[p][row + half * 8]);
            a[mi].q[1] = *reinterpret_cast<const U4*>(&As[p][row + 16 + half * 8]);
        }
        #pragma unroll
        for (int ni = 0; ni < 2; ++ni) {
            int row = (wn * 32 + ni * 16 + lrow) * LDSS;
            b[ni].q[0] = *reinterpret_cast<const U4*>(&Bs[p][row + half * 8]);
            b[ni].q[1] = *reinterpret_cast<const U4*>(&Bs[p][row + 16 + half * 8]);
        }
        #pragma unroll
        for (int mi = 0; mi < 2; ++mi)
            #pragma unroll
            for (int ni = 0; ni < 2; ++ni)
                acc[mi][ni] = __builtin_amdgcn_wmma_f32_16x16x32_f16(
                    false, a[mi].v, false, b[ni].v, (short)0, acc[mi][ni], false, false);
    }

    #pragma unroll
    for (int mi = 0; mi < 2; ++mi)
        #pragma unroll
        for (int ni = 0; ni < 2; ++ni)
            #pragma unroll
            for (int i = 0; i < 8; ++i) {
                int m = mBase + wm * 32 + mi * 16 + half * 8 + i;
                int n = nBase + wn * 32 + ni * 16 + lrow;
                size_t idx = (size_t)m * ldc + n;
                float r = resid ? resid[idx] : 0.0f;
                C[idx] = r + acc[mi][ni][i];
            }
}

// ---------------------------------------------------------------------------
// Simple wave-per-tile WMMA GEMM (used for the narrow N=48 x_proj).
// ---------------------------------------------------------------------------
__global__ __launch_bounds__(32)
void k_gemm_wmma(const _Float16* __restrict__ A, int lda,
                 const _Float16* __restrict__ W, int ldb,
                 float* __restrict__ C, int ldc,
                 const float* __restrict__ resid,
                 int Ktot)
{
    const int lane = threadIdx.x & 31;
    const int half = lane >> 4;
    const int lrow = lane & 15;
    const int n0 = blockIdx.x * 16;
    const int m0 = blockIdx.y * 16;

    v8f acc = {};
    const _Float16* arow = A + (size_t)(m0 + lrow) * lda;
    const _Float16* brow = W + (size_t)(n0 + lrow) * ldb;

    for (int kb = 0; kb < Ktot; kb += 32) {
        union { uint32_t u[8]; v16h v; } af, bf;
        #pragma unroll
        for (int i = 0; i < 8; ++i) {
            int kbase = ((i & 4) ? 16 : 0) + half * 8 + (i & 3) * 2;
            af.u[i] = *reinterpret_cast<const uint32_t*>(arow + kb + kbase);
        }
        #pragma unroll
        for (int i = 0; i < 8; ++i)
            bf.u[i] = *reinterpret_cast<const uint32_t*>(brow + kb + ((i & 4) ? 16 : 0) + half * 8 + (i & 3) * 2);
        acc = __builtin_amdgcn_wmma_f32_16x16x32_f16(false, af.v, false, bf.v,
                                                     (short)0, acc, false, false);
    }
    #pragma unroll
    for (int i = 0; i < 8; ++i) {
        int m = m0 + half * 8 + i;
        size_t idx = (size_t)m * ldc + (n0 + lrow);
        float r = resid ? resid[idx] : 0.0f;
        C[idx] = r + acc[i];
    }
}

// ---------------------------------------------------------------------------
// Causal depthwise conv (K=4) over L + SiLU; reads xin half of xz.
// ---------------------------------------------------------------------------
__global__ void k_conv_silu(const float* __restrict__ xz,
                            const float* __restrict__ cw,
                            const float* __restrict__ cb,
                            float* __restrict__ xc,
                            _Float16* __restrict__ xch)
{
    int idx = blockIdx.x * blockDim.x + threadIdx.x;
    if (idx >= Mrows * DI) return;
    int d  = idx % DI;
    int ml = idx / DI;
    int l  = ml % Lsz;
    int bL = ml - l;
    float acc = cb[d];
    #pragma unroll
    for (int k = 0; k < Kc; ++k) {
        int ll = l - (Kc - 1) + k;
        if (ll >= 0)
            acc += xz[(size_t)(bL + ll) * (2 * DI) + d] * cw[d * Kc + k];
    }
    float s = acc / (1.0f + expf(-acc));   // silu
    xc[idx]  = s;
    xch[idx] = (_Float16)s;
}

// ---------------------------------------------------------------------------
// delta = softplus(dt @ Wdt^T + bdt); dt = dbl[:, 0:16]
// ---------------------------------------------------------------------------
__global__ void k_dt_softplus(const float* __restrict__ dbl,
                              const float* __restrict__ Wdt,
                              const float* __restrict__ bdt,
                              float* __restrict__ delta)
{
    int idx = blockIdx.x * blockDim.x + threadIdx.x;
    if (idx >= Mrows * DI) return;
    int d  = idx % DI;
    int ml = idx / DI;
    float s = bdt[d];
    const float* dt = dbl + (size_t)ml * NPROJ;
    const float* w  = Wdt + (size_t)d * Rr;
    #pragma unroll
    for (int r = 0; r < Rr; ++r) s += dt[r] * w[r];
    delta[idx] = (s > 20.0f) ? s : log1pf(expf(s));
}

// ---------------------------------------------------------------------------
// Selective scan: thread = (b, d), 16 states in registers; B/C rows in LDS.
// dly holds delta on entry and is overwritten in place with y (same index).
// ---------------------------------------------------------------------------
__global__ __launch_bounds__(128)
void k_scan(const float* __restrict__ dbl,
            const float* __restrict__ xc,
            const float* __restrict__ A_log,
            float* __restrict__ dly)
{
    __shared__ float bc[32];             // Bm[0..15], Cm[0..15]
    const int b = blockIdx.x;
    const int d = blockIdx.y * 128 + threadIdx.x;
    float Arow[NST], st[NST];
    #pragma unroll
    for (int n = 0; n < NST; ++n) {
        Arow[n] = -expf(A_log[(size_t)d * NST + n]);
        st[n] = 0.0f;
    }
    for (int l = 0; l < Lsz; ++l) {
        __syncthreads();
        if (threadIdx.x < 32)
            bc[threadIdx.x] = dbl[(size_t)(b * Lsz + l) * NPROJ + Rr + threadIdx.x];
        __syncthreads();
        size_t idx = (size_t)(b * Lsz + l) * DI + d;
        float dlt = dly[idx];
        float du  = dlt * xc[idx];
        float y = 0.0f;
        #pragma unroll
        for (int n = 0; n < NST; ++n) {
            float dA = expf(dlt * Arow[n]);
            st[n] = dA * st[n] + du * bc[n];
            y += st[n] * bc[16 + n];
        }
        dly[idx] = y;
    }
}

// ---------------------------------------------------------------------------
// y = (ys + Dsk*xc) * silu(z); emit f16 for out_proj WMMA GEMM.
// ---------------------------------------------------------------------------
__global__ void k_gate(const float* __restrict__ dly,
                       const float* __restrict__ xc,
                       const float* __restrict__ xz,
                       const float* __restrict__ Dsk,
                       _Float16* __restrict__ yh)
{
    int idx = blockIdx.x * blockDim.x + threadIdx.x;
    if (idx >= Mrows * DI) return;
    int d  = idx % DI;
    int ml = idx / DI;
    float z = xz[(size_t)ml * (2 * DI) + DI + d];
    float y = (dly[idx] + Dsk[d] * xc[idx]) * (z / (1.0f + expf(-z)));
    yh[idx] = (_Float16)y;
}

// ---------------------------------------------------------------------------
// Mean-pool over L + 2-layer head. One block per batch element.
// ---------------------------------------------------------------------------
__global__ __launch_bounds__(256)
void k_pool_head(const float* __restrict__ h,
                 const float* __restrict__ w1, const float* __restrict__ b1,
                 const float* __restrict__ w2, const float* __restrict__ b2,
                 float* __restrict__ out)
{
    __shared__ float pooled[Dsz];
    __shared__ float hh[Hh];
    const int b = blockIdx.x, t = threadIdx.x;
    float s = 0.0f;
    for (int l = 0; l < Lsz; ++l)
        s += h[(size_t)(b * Lsz + l) * Dsz + t];
    pooled[t] = s * (1.0f / Lsz);
    __syncthreads();
    if (t < Hh) {
        float a = b1[t];
        for (int d = 0; d < Dsz; ++d) a += pooled[d] * w1[t * Dsz + d];
        hh[t] = a > 0.0f ? a : 0.0f;
    }
    __syncthreads();
    if (t == 0) {
        float o = b2[0];
        for (int j = 0; j < Hh; ++j) o += hh[j] * w2[j];
        out[b] = o;
    }
}

// ---------------------------------------------------------------------------
extern "C" void kernel_launch(void* const* d_in, const int* in_sizes, int n_in,
                              void* d_out, int out_size, void* d_ws, size_t ws_size,
                              hipStream_t stream)
{
    (void)in_sizes; (void)n_in; (void)out_size; (void)ws_size;
    const float* x    = (const float*)d_in[0];
    const float* ipw  = (const float*)d_in[1];
    const float* ipb  = (const float*)d_in[2];
    const float* ln_g = (const float*)d_in[3];
    const float* ln_b = (const float*)d_in[4];
    const float* Wi   = (const float*)d_in[5];
    const float* cw   = (const float*)d_in[6];
    const float* cb   = (const float*)d_in[7];
    const float* Wx   = (const float*)d_in[8];
    const float* Wdt  = (const float*)d_in[9];
    const float* bdt  = (const float*)d_in[10];
    const float* Alog = (const float*)d_in[11];
    const float* Dsk  = (const float*)d_in[12];
    const float* Wo   = (const float*)d_in[13];
    const float* hw1  = (const float*)d_in[14];
    const float* hb1  = (const float*)d_in[15];
    const float* hw2  = (const float*)d_in[16];
    const float* hb2  = (const float*)d_in[17];
    float* out = (float*)d_out;

    char* wsb = (char*)d_ws;
    size_t off = 0;
    auto take = [&](size_t bytes) -> char* {
        char* p = wsb + off;
        off = (off + bytes + 255) & ~(size_t)255;
        return p;
    };
    float*    h    = (float*)   take((size_t)Mrows * Dsz * 4);
    float*    xz   = (float*)   take((size_t)Mrows * 2 * DI * 4);
    float*    xc   = (float*)   take((size_t)Mrows * DI * 4);
    float*    dly  = (float*)   take((size_t)Mrows * DI * 4);     // delta, reused as y
    float*    dbl  = (float*)   take((size_t)Mrows * NPROJ * 4);
    _Float16* xnh  = (_Float16*)take((size_t)Mrows * Dsz * 2);
    _Float16* xch  = (_Float16*)take((size_t)Mrows * DI * 2);
    _Float16* yh   = (_Float16*)take((size_t)Mrows * DI * 2);
    _Float16* Wih  = (_Float16*)take((size_t)NL * 2 * DI * Dsz * 2);
    _Float16* Wxh  = (_Float16*)take((size_t)NL * NPROJ * DI * 2);
    _Float16* Woh  = (_Float16*)take((size_t)NL * Dsz * DI * 2);

    // stage 0: input projection + f16 weight mirrors
    {
        int n = Mrows * Dsz;
        k_input_proj<<<(n + 255) / 256, 256, 0, stream>>>(x, ipw, ipb, h);
    }
    { int n = NL * 2 * DI * Dsz; k_f32_to_f16<<<(n + 255) / 256, 256, 0, stream>>>(Wi, Wih, n); }
    { int n = NL * NPROJ * DI;   k_f32_to_f16<<<(n + 255) / 256, 256, 0, stream>>>(Wx, Wxh, n); }
    { int n = NL * Dsz * DI;     k_f32_to_f16<<<(n + 255) / 256, 256, 0, stream>>>(Wo, Woh, n); }

    for (int layer = 0; layer < NL; ++layer) {
        k_layernorm<<<Mrows, 256, 0, stream>>>(h, ln_g + layer * Dsz, ln_b + layer * Dsz, xnh);

        // in_proj: xz[M,1024] = xn[M,256] * Wi[1024,256]^T   (LDS-tiled WMMA)
        k_gemm_wmma_tiled<<<dim3(2 * DI / 128, Mrows / 64), 256, 0, stream>>>(
            xnh, Dsz, Wih + (size_t)layer * 2 * DI * Dsz, Dsz, xz, 2 * DI, nullptr, Dsz);

        {
            int n = Mrows * DI;
            k_conv_silu<<<(n + 255) / 256, 256, 0, stream>>>(
                xz, cw + (size_t)layer * DI * Kc, cb + (size_t)layer * DI, xc, xch);
        }

        // x_proj: dbl[M,48] = xc[M,512] * Wx[48,512]^T   (narrow N -> simple WMMA)
        k_gemm_wmma<<<dim3(NPROJ / 16, Mrows / 16), 32, 0, stream>>>(
            xch, DI, Wxh + (size_t)layer * NPROJ * DI, DI, dbl, NPROJ, nullptr, DI);

        {
            int n = Mrows * DI;
            k_dt_softplus<<<(n + 255) / 256, 256, 0, stream>>>(
                dbl, Wdt + (size_t)layer * DI * Rr, bdt + (size_t)layer * DI, dly);
        }

        k_scan<<<dim3(Bsz, DI / 128), 128, 0, stream>>>(
            dbl, xc, Alog + (size_t)layer * DI * NST, dly);

        {
            int n = Mrows * DI;
            k_gate<<<(n + 255) / 256, 256, 0, stream>>>(
                dly, xc, xz, Dsk + (size_t)layer * DI, yh);
        }

        // out_proj + residual: h[M,256] = h + y[M,512] * Wo[256,512]^T  (LDS-tiled)
        k_gemm_wmma_tiled<<<dim3(Dsz / 128, Mrows / 64), 256, 0, stream>>>(
            yh, DI, Woh + (size_t)layer * Dsz * DI, DI, h, Dsz, h, DI);
    }

    k_pool_head<<<Bsz, 256, 0, stream>>>(h, hw1, hb1, hw2, hb2, out);
}
